// SpMiddleNoDownsampleXYSingleFrame_21139829031359
// MI455X (gfx1250) — matrix operand order repaired and checked by reading, not compiled, not run on hardware
//
#include <hip/hip_runtime.h>
#include <hip/hip_bf16.h>

// ---------------------------------------------------------------------------
// Types for WMMA (CDNA5 gfx1250, wave32)
// ---------------------------------------------------------------------------
typedef __attribute__((ext_vector_type(16))) __bf16        v16bf;
typedef __attribute__((ext_vector_type(8)))  float         v8f;
typedef __attribute__((ext_vector_type(4)))  unsigned int  v4u;
typedef __attribute__((ext_vector_type(8)))  unsigned int  v8u;

__device__ __forceinline__ unsigned short f2bf(float f) {
  unsigned int u = __float_as_uint(f);
  unsigned int r = u + 0x7FFFu + ((u >> 16) & 1u);   // round to nearest even
  return (unsigned short)(r >> 16);
}

__device__ __forceinline__ v16bf pack_ab(v4u lo, v4u hi) {
  v8u c = __builtin_shufflevector(lo, hi, 0, 1, 2, 3, 4, 5, 6, 7);
  return __builtin_bit_cast(v16bf, c);
}

// ---------------------------------------------------------------------------
// Fill / scatter / mask kernels
// ---------------------------------------------------------------------------
__global__ void fill_u32_kernel(unsigned int* p, long n, unsigned int v) {
  long t = (long)blockIdx.x * blockDim.x + threadIdx.x;
  if (t < n) p[t] = v;
}

__global__ void scatter_kernel(const float* __restrict__ feats,
                               const int* __restrict__ coords,
                               unsigned short* __restrict__ grid,
                               unsigned char* __restrict__ mask, int N) {
  int t = blockIdx.x * blockDim.x + threadIdx.x;
  if (t >= N * 16) return;
  int p = t >> 4, c = t & 15;
  int z = coords[p * 4 + 1];
  int y = coords[p * 4 + 2];
  int x = coords[p * 4 + 3];
  long sp = ((long)z * 128 + y) * 128 + x;
  grid[sp * 16 + c] = f2bf(feats[t]);
  if (c == 0) mask[sp] = 1;
}

struct MaskP {
  const unsigned char* in;
  unsigned char* out;
  int Din, Dout, H, W, KD, KH, KW, sd, pd, ph, pw;
  long total;
};

__global__ void mask_down_kernel(MaskP P) {
  long t = (long)blockIdx.x * blockDim.x + threadIdx.x;
  if (t >= P.total) return;
  int w = (int)(t % P.W);
  long r = t / P.W;
  int h = (int)(r % P.H);
  int d = (int)(r / P.H);
  unsigned char any = 0;
  for (int kd = 0; kd < P.KD; ++kd) {
    int din = d * P.sd + kd - P.pd;
    if (din < 0 || din >= P.Din) continue;
    for (int kh = 0; kh < P.KH; ++kh) {
      int hin = h + kh - P.ph;
      if (hin < 0 || hin >= P.H) continue;
      for (int kw = 0; kw < P.KW; ++kw) {
        int win = w + kw - P.pw;
        if (win < 0 || win >= P.W) continue;
        any |= P.in[((long)din * P.H + hin) * P.W + win];
      }
    }
  }
  P.out[t] = any ? 1 : 0;
}

// ---------------------------------------------------------------------------
// Weight repack: DHWIO fp32 -> per-(tap,kblock,nblock) 1KB WMMA-B blocks.
// Block = 32 lanes x 16 bf16 (lane-major).  Lane half h, vgpr j, elem p:
//   K = 8*h + 2*j + p + (j>=4 ? 8 : 0), N = lane&15  -> bf16 at dst[blk*512 + lane*16 + 2j+p]
// ---------------------------------------------------------------------------
struct RepP {
  const float* w;
  unsigned short* dst;
  int KD, KH, KW, IC, OC, NK, NB, nBlocks;
};

__global__ void repack_kernel(RepP P) {
  int t = blockIdx.x * blockDim.x + threadIdx.x;
  if (t >= P.nBlocks * 32) return;
  int lane = t & 31;
  int blk  = t >> 5;
  int nb = blk % P.NB;
  int r2 = blk / P.NB;
  int kb = r2 % P.NK;
  int tap = r2 / P.NK;
  int kw = tap % P.KW;
  int r3 = tap / P.KW;
  int kh = r3 % P.KH;
  int kd = r3 / P.KH;
  int n = nb * 16 + (lane & 15);
  int half = lane >> 4;
  const float* wsrc = P.w + (long)((kd * P.KH + kh) * P.KW + kw) * P.IC * P.OC;
  unsigned short* dp = P.dst + (long)blk * 512 + lane * 16;
#pragma unroll
  for (int j = 0; j < 8; ++j) {
#pragma unroll
    for (int p = 0; p < 2; ++p) {
      int k = 8 * half + 2 * j + p + (j >= 4 ? 8 : 0);
      int c = kb * 32 + k;
      float v = (c < P.IC) ? wsrc[(long)c * P.OC + n] : 0.f;
      dp[2 * j + p] = f2bf(v);
    }
  }
}

// ---------------------------------------------------------------------------
// Implicit-GEMM conv3d + BN + ReLU + mask, WMMA f32 = bf16 x bf16 + f32.
// One wave per (d, h, 16-wide w-tile); the wave computes ALL NB 16-channel
// output blocks, so each A fragment is reused by NB WMMAs.
// grid = (W/16, H, Dout)  -> no integer division in the kernel.
// All fragments are native ext-vectors (no unions) -> fully register-resident.
// ---------------------------------------------------------------------------
struct ConvP {
  const unsigned short* in;     // bf16 NDHWC
  const unsigned short* wrep;   // repacked B blocks
  const float* scale;
  const float* bias;
  const unsigned char* mask;    // at output resolution
  float* outf;
  unsigned short* outh;
  int Din, Dout, H, W, IC, OC, KD, KH, KW, sd, pd, ph, pw, NK, out_f32;
};

template <int NB>
__global__ __launch_bounds__(32) void conv_wmma_kernel(ConvP P) {
  const int lane = threadIdx.x;
  const int wt = blockIdx.x;
  const int h  = blockIdx.y;
  const int d  = blockIdx.z;
  const int half = lane >> 4;
  const int n16  = lane & 15;
  const int w0 = wt << 4;
  const v4u z4 = (v4u)0u;

  v8f acc[NB];
#pragma unroll
  for (int i = 0; i < NB; ++i) acc[i] = (v8f)0.f;

  const unsigned short* bpLane = P.wrep + (lane << 4);

  for (int kd = 0; kd < P.KD; ++kd) {
    const int din = d * P.sd + kd - P.pd;
    if (din < 0 || din >= P.Din) continue;          // uniform: EXEC stays all-1s
    for (int kh = 0; kh < P.KH; ++kh) {
      const int hin = h + kh - P.ph;
      if (hin < 0 || hin >= P.H) continue;          // uniform
      for (int kw = 0; kw < P.KW; ++kw) {
        const int tap = (kd * P.KH + kh) * P.KW + kw;
        const int win = w0 + n16 + kw - P.pw;       // per-lane row address
        const bool wok = (win >= 0) && (win < P.W);
        const unsigned short* inrow =
            P.in + (((long)din * P.H + hin) * P.W + win) * P.IC;
        for (int kb = 0; kb < P.NK; ++kb) {
          const int cA = kb * 32 + half * 8;        // K {0..7}/{8..15} chunk
          const int cB = cA + 16;                   // K {16..23}/{24..31} chunk
          v4u alo = z4, ahi = z4;
          if (wok && cA < P.IC) alo = *(const v4u*)(inrow + cA);
          if (wok && cB < P.IC) ahi = *(const v4u*)(inrow + cB);
          const v16bf av = pack_ab(alo, ahi);
          const unsigned short* bp =
              bpLane + ((long)((tap * P.NK + kb) * NB) << 9);
#pragma unroll
          for (int nb = 0; nb < NB; ++nb) {
            v4u blo = *(const v4u*)(bp + (nb << 9) + 0);
            v4u bhi = *(const v4u*)(bp + (nb << 9) + 8);
            const v16bf bv = pack_ab(blo, bhi);
            acc[nb] = __builtin_amdgcn_wmma_f32_16x16x32_bf16(
                false, av, false, bv, (short)0, acc[nb], false, false);
          }
        }
      }
    }
  }

  // Epilogue: C layout -> lane n16 = N (out channel), VGPR r -> M = r + 8*half.
  const long spBase = ((long)d * P.H + h) * P.W + w0;
#pragma unroll
  for (int nb = 0; nb < NB; ++nb) {
    const int n = nb * 16 + n16;
    const float sc = P.scale[n];
    const float bi = P.bias[n];
#pragma unroll
    for (int r = 0; r < 8; ++r) {
      const int M = r + 8 * half;
      const long sp = spBase + M;
      float v = acc[nb][r] * sc + bi;
      v = v > 0.f ? v : 0.f;
      v *= (float)P.mask[sp];
      const long o = sp * P.OC + n;
      if (P.out_f32) P.outf[o] = v;
      else           P.outh[o] = f2bf(v);
    }
  }
}

// ---------------------------------------------------------------------------
// Host-side orchestration
// ---------------------------------------------------------------------------
struct LD { int KD, KH, KW, IC, OC, Din, Dout, sd, pd, ph, pw, mask; };

extern "C" void kernel_launch(void* const* d_in, const int* in_sizes, int n_in,
                              void* d_out, int out_size, void* d_ws, size_t ws_size,
                              hipStream_t stream) {
  const int Hh = 128, Ww = 128;
  const float* feats = (const float*)d_in[0];
  const int* coords  = (const int*)d_in[1];
  const int N = in_sizes[0] / 16;

  char* ws = (char*)d_ws;
  const size_t OFF_ACT0  = 0;
  const size_t OFF_ACT1  = 45ull << 20;
  const size_t OFF_MSK0  = 90ull << 20;
  const size_t OFF_MSK1  = OFF_MSK0 + (768u << 10);
  const size_t OFF_MSK2  = OFF_MSK1 + (512u << 10);
  const size_t OFF_MSK3  = OFF_MSK2 + (256u << 10);
  const size_t OFF_WREP  = OFF_MSK3 + (256u << 10);

  unsigned short* grid0 = (unsigned short*)(ws + OFF_ACT0);
  unsigned char* mptr[4] = {(unsigned char*)(ws + OFF_MSK0),
                            (unsigned char*)(ws + OFF_MSK1),
                            (unsigned char*)(ws + OFF_MSK2),
                            (unsigned char*)(ws + OFF_MSK3)};

  // 1) zero the scatter grid (41x128x128x16 bf16) and mask0 (41x128x128 u8)
  {
    long n = 41L * 128 * 128 * 16 * 2 / 4;
    fill_u32_kernel<<<(unsigned)((n + 255) / 256), 256, 0, stream>>>(
        (unsigned int*)grid0, n, 0u);
    long nm = 41L * 128 * 128 / 4;
    fill_u32_kernel<<<(unsigned)((nm + 255) / 256), 256, 0, stream>>>(
        (unsigned int*)mptr[0], nm, 0u);
  }

  // 2) scatter voxel features + mask0
  scatter_kernel<<<(N * 16 + 255) / 256, 256, 0, stream>>>(feats, coords, grid0,
                                                           mptr[0], N);

  // 3) occupancy mask pyramid
  {
    MaskP mp{};
    mp.in = mptr[0]; mp.out = mptr[1];
    mp.Din = 41; mp.Dout = 21; mp.H = Hh; mp.W = Ww;
    mp.KD = 3; mp.KH = 3; mp.KW = 3; mp.sd = 2; mp.pd = 1; mp.ph = 1; mp.pw = 1;
    mp.total = 21L * Hh * Ww;
    mask_down_kernel<<<(unsigned)((mp.total + 255) / 256), 256, 0, stream>>>(mp);

    mp.in = mptr[1]; mp.out = mptr[2];
    mp.Din = 21; mp.Dout = 10; mp.pd = 0; mp.total = 10L * Hh * Ww;
    mask_down_kernel<<<(unsigned)((mp.total + 255) / 256), 256, 0, stream>>>(mp);

    mp.in = mptr[2]; mp.out = mptr[3];
    mp.Din = 10; mp.Dout = 4; mp.KH = 1; mp.KW = 1; mp.ph = 0; mp.pw = 0;
    mp.total = 4L * Hh * Ww;
    mask_down_kernel<<<(unsigned)((mp.total + 255) / 256), 256, 0, stream>>>(mp);
  }

  static const LD L[11] = {
      {3, 3, 3, 16, 32, 41, 41, 1, 1, 1, 1, 0},
      {3, 3, 3, 32, 32, 41, 41, 1, 1, 1, 1, 0},
      {3, 3, 3, 32, 64, 41, 21, 2, 1, 1, 1, 1},
      {3, 3, 3, 64, 64, 21, 21, 1, 1, 1, 1, 1},
      {3, 3, 3, 64, 64, 21, 21, 1, 1, 1, 1, 1},
      {3, 3, 3, 64, 64, 21, 21, 1, 1, 1, 1, 1},
      {3, 3, 3, 64, 64, 21, 10, 2, 0, 1, 1, 2},
      {3, 3, 3, 64, 64, 10, 10, 1, 1, 1, 1, 2},
      {3, 3, 3, 64, 64, 10, 10, 1, 1, 1, 1, 2},
      {3, 3, 3, 64, 64, 10, 10, 1, 1, 1, 1, 2},
      {3, 1, 1, 64, 64, 10, 4, 2, 0, 0, 0, 3},
  };

  // 4) repack all weights into WMMA-B layout
  size_t wrepOff[12];
  {
    size_t off = OFF_WREP;
    for (int i = 0; i < 11; ++i) {
      wrepOff[i] = off;
      int NK = (L[i].IC + 31) / 32;
      int NB = L[i].OC / 16;
      int nblk = L[i].KD * L[i].KH * L[i].KW * NK * NB;
      RepP rp{};
      rp.w = (const float*)d_in[2 + 3 * i];
      rp.dst = (unsigned short*)(ws + off);
      rp.KD = L[i].KD; rp.KH = L[i].KH; rp.KW = L[i].KW;
      rp.IC = L[i].IC; rp.OC = L[i].OC; rp.NK = NK; rp.NB = NB; rp.nBlocks = nblk;
      int thr = nblk * 32;
      repack_kernel<<<(thr + 255) / 256, 256, 0, stream>>>(rp);
      off += (size_t)nblk * 1024;
    }
    wrepOff[11] = off;
  }

  // 5) conv layers (ping-pong bf16 activations; last layer -> d_out fp32)
  for (int i = 0; i < 11; ++i) {
    int NK = (L[i].IC + 31) / 32;
    int NB = L[i].OC / 16;
    ConvP cp{};
    cp.in = (i == 0) ? grid0
                     : (unsigned short*)(ws + (((i % 2) == 1) ? OFF_ACT1 : OFF_ACT0));
    cp.wrep = (const unsigned short*)(ws + wrepOff[i]);
    cp.scale = (const float*)d_in[3 + 3 * i];
    cp.bias  = (const float*)d_in[4 + 3 * i];
    cp.mask  = mptr[L[i].mask];
    if (i == 10) { cp.outf = (float*)d_out; cp.outh = nullptr; cp.out_f32 = 1; }
    else {
      cp.outh = (unsigned short*)(ws + (((i % 2) == 0) ? OFF_ACT1 : OFF_ACT0));
      cp.outf = nullptr; cp.out_f32 = 0;
    }
    cp.Din = L[i].Din; cp.Dout = L[i].Dout; cp.H = Hh; cp.W = Ww;
    cp.IC = L[i].IC; cp.OC = L[i].OC;
    cp.KD = L[i].KD; cp.KH = L[i].KH; cp.KW = L[i].KW;
    cp.sd = L[i].sd; cp.pd = L[i].pd; cp.ph = L[i].ph; cp.pw = L[i].pw;
    cp.NK = NK; cp.out_f32 = (i == 10) ? 1 : 0;

    dim3 grid(Ww / 16, Hh, L[i].Dout);
    if (NB == 2) conv_wmma_kernel<2><<<grid, 32, 0, stream>>>(cp);
    else         conv_wmma_kernel<4><<<grid, 32, 0, stream>>>(cp);
  }
}
